// TopKSAE_29008209117482
// MI455X (gfx1250) — matrix operand order repaired and checked by reading, not compiled, not run on hardware
//
#include <hip/hip_runtime.h>
#include <hip/hip_bf16.h>

// ---------------------------------------------------------------------------
// TopK SAE for MI455X (gfx1250):
//   bf16 WMMA encode with double-buffered TDM (tensor_load_to_lds) staging ->
//   LDS radix top-k -> sparse decode. Matrix math on v_wmma_f32_16x16x32_bf16.
// ---------------------------------------------------------------------------

#define D_MODEL 2048
#define D_SAE   32768
#define N_TOK   8192
#define TOPK    64
#define KSTEPS  (D_MODEL / 32)

typedef __attribute__((ext_vector_type(16))) __bf16 v16bf;
typedef __attribute__((ext_vector_type(8)))  float  v8f;
typedef unsigned int u32x4v __attribute__((ext_vector_type(4)));
typedef int          i32x4v __attribute__((ext_vector_type(4)));
typedef int          i32x8v __attribute__((ext_vector_type(8)));

union Frag16 {
    uint4  q[2];
    v16bf  v;
};

#if defined(__has_builtin)
#if __has_builtin(__builtin_amdgcn_tensor_load_to_lds)
#define HAVE_TDM 1
#endif
#endif
#ifndef HAVE_TDM
#define HAVE_TDM 0
#endif

static __device__ __forceinline__ unsigned short f2bf(float f) {
    unsigned u = __float_as_uint(f);
    unsigned r = (u + 0x7FFFu + ((u >> 16) & 1u)) >> 16;   // RNE
    return (unsigned short)r;
}

// ------------------------- prep: xc = bf16(x - b_pre) ----------------------
__global__ void sae_prep_x(const float* __restrict__ x,
                           const float* __restrict__ b_pre,
                           unsigned short* __restrict__ xc) {
    int i = blockIdx.x * blockDim.x + threadIdx.x;
    int stride = gridDim.x * blockDim.x;
    for (; i < N_TOK * D_MODEL; i += stride)
        xc[i] = f2bf(x[i] - b_pre[i & (D_MODEL - 1)]);
}

// ------------------------- prep: W_enc -> bf16 -----------------------------
__global__ void sae_prep_w(const float* __restrict__ W,
                           unsigned short* __restrict__ Wb, int total) {
    int i = blockIdx.x * blockDim.x + threadIdx.x;
    int stride = gridDim.x * blockDim.x;
    for (; i < total; i += stride)
        Wb[i] = f2bf(W[i]);
}

// ------------------- prep: W_dec [D_MODEL][D_SAE] -> WdT [D_SAE][D_MODEL] --
__global__ void sae_transpose_wdec(const float* __restrict__ Wdec,
                                   float* __restrict__ WdT) {
    __shared__ float tile[32][33];
    const int s0 = blockIdx.x * 32;
    const int d0 = blockIdx.y * 32;
    const int c = threadIdx.x & 31;
    const int r = threadIdx.x >> 5;            // 0..7
#pragma unroll
    for (int j = 0; j < 32; j += 8)
        tile[r + j][c] = Wdec[(size_t)(d0 + r + j) * D_SAE + (s0 + c)];
    __syncthreads();
#pragma unroll
    for (int j = 0; j < 32; j += 8)
        WdT[(size_t)(s0 + r + j) * D_MODEL + (d0 + c)] = tile[c][r + j];
}

// ------------------- encoder GEMM (bf16 WMMA + double-buffered TDM) --------
// z[n0:n0+128, s0:s0+128] = xc @ Wenc^T + b_enc, fp32 accumulate.
// 256 threads = 8 waves; wave (rw = w&3, cw = w>>2) owns 32 rows x 64 cols
// (2x4 accumulators of 16x16). K staged in slabs of 32 through 2x-buffered
// LDS: wave0/wave1 DMA slab i+1 with tensor_load_to_lds, then wait
// TENSORcnt<=1 (slab i landed, slab i+1 overlaps the 8 WMMAs of slab i).
// K loop manually unrolled x2 so the buffer index is a literal and the
// accumulators stay register-stable (no v_mov patch-up copies).
__global__ __launch_bounds__(256)
void sae_encode(const unsigned short* __restrict__ xc,
                const unsigned short* __restrict__ Wb,
                const float* __restrict__ b_enc,
                float* __restrict__ zout) {
    __shared__ __align__(16) unsigned short Abuf[2][128 * 32];   // 2 x 8KB
    __shared__ __align__(16) unsigned short Bbuf[2][128 * 32];   // 2 x 8KB

    const int t    = threadIdx.x;
    const int lane = t & 31;
    const int w    = t >> 5;
    const int rw   = w & 3;          // 32-row group 0..3
    const int cw   = w >> 2;         // 64-col group 0..1

    const int s0 = blockIdx.x * 128;
    const int n0 = blockIdx.y * 128;

    const int half = lane >> 4;      // fragment K-half selector
    const int l16  = lane & 15;

    v8f acc[2][4];
    const v8f vzero = {};
#pragma unroll
    for (int mt = 0; mt < 2; ++mt)
#pragma unroll
        for (int ct = 0; ct < 4; ++ct) acc[mt][ct] = vzero;

    auto computeSlab = [&](int cur) {
        const unsigned short* Ab = &Abuf[cur][0];
        const unsigned short* Bb = &Bbuf[cur][0];
        // A fragments (16x32 bf16): lane-half selects K chunks {0..7,16..23}/{8..15,24..31}
        Frag16 fa[2];
#pragma unroll
        for (int mt = 0; mt < 2; ++mt) {
            const int aBase = (rw * 32 + mt * 16 + l16) * 32;
            fa[mt].q[0] = *(const uint4*)&Ab[aBase + half * 8];
            fa[mt].q[1] = *(const uint4*)&Ab[aBase + 16 + half * 8];
        }
        // B fed as Wenc rows (== B^T): lane-half selects K chunks {0..7,16..23}/{8..15,24..31}
#pragma unroll
        for (int ct = 0; ct < 4; ++ct) {
            const int bBase = (cw * 64 + ct * 16 + l16) * 32;
            Frag16 fb;
            fb.q[0] = *(const uint4*)&Bb[bBase + half * 8];
            fb.q[1] = *(const uint4*)&Bb[bBase + 16 + half * 8];
            acc[0][ct] = __builtin_amdgcn_wmma_f32_16x16x32_bf16(
                             false, fa[0].v, false, fb.v, (short)0, acc[0][ct], false, false);
            acc[1][ct] = __builtin_amdgcn_wmma_f32_16x16x32_bf16(
                             false, fa[1].v, false, fb.v, (short)0, acc[1][ct], false, false);
        }
    };

#if HAVE_TDM
    // ---- Tensor DMA descriptors (CDNA5 ISA) ----
    unsigned long long gA = (unsigned long long)(uintptr_t)xc +
                            2ull * ((size_t)n0 * D_MODEL);
    unsigned long long gB = (unsigned long long)(uintptr_t)Wb +
                            2ull * ((size_t)s0 * D_MODEL);
    const unsigned ldsA = (unsigned)(uintptr_t)(void*)&Abuf[0][0];
    const unsigned ldsB = (unsigned)(uintptr_t)(void*)&Bbuf[0][0];

    // group0: [1:0]=count=1 | [63:32]=lds_addr | [120:64]=global_addr | [127:126]=type=2
    u32x4v g0A, g0B;
    g0A.x = 1u; g0B.x = 1u;

    // group1: data_size=1(2B); tensor_dim0=D_MODEL; tensor_dim1=rows;
    //         tile_dim0=32; tile_dim1=128; tensor_dim0_stride=D_MODEL
    i32x8v g1A, g1B;
    {
        const unsigned rowsA = N_TOK, rowsB = D_SAE;
        g1A[0] = (int)(1u << 16);
        g1A[1] = (int)((D_MODEL & 0xFFFFu) << 16);
        g1A[2] = (int)(((D_MODEL >> 16) & 0xFFFFu) | ((rowsA & 0xFFFFu) << 16));
        g1A[3] = (int)(((rowsA >> 16) & 0xFFFFu) | (32u << 16));
        g1A[4] = (int)(128u);
        g1A[5] = (int)(D_MODEL);
        g1A[6] = 0; g1A[7] = 0;
        g1B = g1A;
        g1B[2] = (int)(((D_MODEL >> 16) & 0xFFFFu) | ((rowsB & 0xFFFFu) << 16));
        g1B[3] = (int)(((rowsB >> 16) & 0xFFFFu) | (32u << 16));
    }
    const i32x4v z4 = {};
#if defined(__clang_major__) && (__clang_major__ >= 23)
    const i32x8v z8 = {};
#define TDM_LOAD(G0, G1) __builtin_amdgcn_tensor_load_to_lds((G0), (G1), z4, z4, z8, 0)
#else
#define TDM_LOAD(G0, G1) __builtin_amdgcn_tensor_load_to_lds((G0), (G1), z4, z4, 0)
#endif
    auto issueA = [&](int buf) {
        g0A.y = ldsA + (unsigned)buf * (128u * 32u * 2u);
        g0A.z = (unsigned)(gA & 0xFFFFFFFFull);
        g0A.w = (unsigned)((gA >> 32) & 0x1FFFFFFull) | 0x80000000u;   // type=2
        TDM_LOAD(g0A, g1A);
        gA += 64;                                // 32 bf16 = 64B per K-slab
    };
    auto issueB = [&](int buf) {
        g0B.y = ldsB + (unsigned)buf * (128u * 32u * 2u);
        g0B.z = (unsigned)(gB & 0xFFFFFFFFull);
        g0B.w = (unsigned)((gB >> 32) & 0x1FFFFFFull) | 0x80000000u;
        TDM_LOAD(g0B, g1B);
        gB += 64;
    };
    // prologue: slab 0 in flight before the loop
    if (w == 0)      issueA(0);
    else if (w == 1) issueB(0);

    for (int it = 0; it < KSTEPS; it += 2) {
        // ---- slab it (buffer 0); DMA slab it+1 into buffer 1 ----
        __syncthreads();                                // buffer 1 consumed
        if (w == 0)      { issueA(1); __builtin_amdgcn_s_wait_tensorcnt(1); }
        else if (w == 1) { issueB(1); __builtin_amdgcn_s_wait_tensorcnt(1); }
        __syncthreads();                                // slab it visible
        computeSlab(0);

        // ---- slab it+1 (buffer 1); DMA slab it+2 into buffer 0 ----
        const bool more = (it + 2) < KSTEPS;
        __syncthreads();                                // buffer 0 consumed
        if (w == 0) {
            if (more) { issueA(0); __builtin_amdgcn_s_wait_tensorcnt(1); }
            else      {            __builtin_amdgcn_s_wait_tensorcnt(0); }
        } else if (w == 1) {
            if (more) { issueB(0); __builtin_amdgcn_s_wait_tensorcnt(1); }
            else      {            __builtin_amdgcn_s_wait_tensorcnt(0); }
        }
        __syncthreads();                                // slab it+1 visible
        computeSlab(1);
    }
#else
    // ---- fallback staging: per-thread b128 global->LDS (buffer 0 only) ----
    const int lrow = t >> 1;               // 0..127
    const int lc   = (t & 1) * 16;         // 0,16
    const unsigned short* pA = xc + (size_t)(n0 + lrow) * D_MODEL + lc;
    const unsigned short* pB = Wb + (size_t)(s0 + lrow) * D_MODEL + lc;
    for (int k0 = 0; k0 < D_MODEL; k0 += 32) {
        uint4 va0 = *(const uint4*)(pA + k0);
        uint4 va1 = *(const uint4*)(pA + k0 + 8);
        uint4 vb0 = *(const uint4*)(pB + k0);
        uint4 vb1 = *(const uint4*)(pB + k0 + 8);
        __syncthreads();
        *(uint4*)&Abuf[0][lrow * 32 + lc]     = va0;
        *(uint4*)&Abuf[0][lrow * 32 + lc + 8] = va1;
        *(uint4*)&Bbuf[0][lrow * 32 + lc]     = vb0;
        *(uint4*)&Bbuf[0][lrow * 32 + lc + 8] = vb1;
        __syncthreads();
        computeSlab(0);
    }
#endif

    // C/D layout: lane fixes column N (= l16, + half*8 rows), VGPR j walks M
    const int mb = half * 8;
#pragma unroll
    for (int ct = 0; ct < 4; ++ct) {
        const int scol = s0 + cw * 64 + ct * 16 + l16;
        const float bias = b_enc[scol];
#pragma unroll
        for (int mt = 0; mt < 2; ++mt) {
#pragma unroll
            for (int j = 0; j < 8; ++j) {
                const int row = n0 + rw * 32 + mt * 16 + mb + j;
                zout[(size_t)row * D_SAE + scol] = acc[mt][ct][j] + bias;
            }
        }
    }
}

// --------------------- exact top-64 + in-place sparsify --------------------
// One WG per token row. Row keys (32768 x u32) live in LDS (128KB < 320KB/WGP).
// 4-round MSB radix refine finds the exact K-th threshold incl. tie handling.
__global__ __launch_bounds__(256)
void sae_topk(float* __restrict__ zbase,
              int* __restrict__ pidx,
              float* __restrict__ pval) {
    extern __shared__ unsigned smem[];
    unsigned* keys = smem;                 // D_SAE entries
    unsigned* hist = smem + D_SAE;         // 256 bins
    __shared__ unsigned s_prefix, s_remaining, s_tie, s_nsel;

    const int n = blockIdx.x;
    const int t = threadIdx.x;
    float* zrow = zbase + (size_t)n * D_SAE;

    for (int i = t; i < D_SAE; i += 256) {
        unsigned u = __float_as_uint(zrow[i]);
        keys[i] = u ^ ((u >> 31) ? 0xFFFFFFFFu : 0x80000000u);  // monotone key
    }
    if (t == 0) { s_prefix = 0u; s_remaining = TOPK; s_tie = 0u; s_nsel = 0u; }
    __syncthreads();

    unsigned maskHigh = 0u;
    for (int r = 0; r < 4; ++r) {
        const int shift = 24 - 8 * r;
        hist[t] = 0u;
        __syncthreads();
        const unsigned prefix = s_prefix;
        for (int i = t; i < D_SAE; i += 256) {
            unsigned k = keys[i];
            if ((k & maskHigh) == prefix)
                atomicAdd(&hist[(k >> shift) & 255u], 1u);
        }
        __syncthreads();
        if (t == 0) {
            unsigned rem = s_remaining, cum = 0u;
            int chosen = 0;
            for (int b = 255; b >= 0; --b) {
                unsigned c = hist[b];
                if (cum + c >= rem) { chosen = b; break; }
                cum += c;
            }
            s_prefix = prefix | ((unsigned)chosen << shift);
            s_remaining = rem - cum;
        }
        maskHigh |= (0xFFu << shift);
        __syncthreads();
    }

    const unsigned T = s_prefix;         // exact K-th largest key
    const unsigned need = s_remaining;   // ties at T to keep

    for (int i = t; i < D_SAE; i += 256) {
        const unsigned k = keys[i];
        bool keep = false;
        if (k > T) keep = true;
        else if (k == T) keep = (atomicAdd(&s_tie, 1u) < need);
        if (keep) {
            const unsigned u = (k & 0x80000000u) ? (k ^ 0x80000000u) : ~k;
            const float v = __uint_as_float(u);
            const unsigned slot = atomicAdd(&s_nsel, 1u);
            if (slot < TOPK) {
                pidx[n * TOPK + slot] = i;
                pval[n * TOPK + slot] = v;
            }
            zrow[i] = v;
        } else {
            zrow[i] = 0.0f;
        }
    }
}

// --------------------------- sparse decode ---------------------------------
// out[n,:] = b_dec + sum_k val_k * WdT[idx_k, :]   (contiguous 8KB row reads)
__global__ __launch_bounds__(256)
void sae_decode(const float* __restrict__ WdT,
                const int* __restrict__ pidx,
                const float* __restrict__ pval,
                const float* __restrict__ b_dec,
                float* __restrict__ out) {
    const int n = blockIdx.x;
    const int t = threadIdx.x;
    float acc[8];
#pragma unroll
    for (int j = 0; j < 8; ++j) acc[j] = b_dec[j * 256 + t];

    for (int k = 0; k < TOPK; ++k) {
        const int idx = pidx[n * TOPK + k];
        const float v = pval[n * TOPK + k];
        const float* wr = WdT + (size_t)idx * D_MODEL;
#pragma unroll
        for (int j = 0; j < 8; ++j) acc[j] += v * wr[j * 256 + t];
    }
#pragma unroll
    for (int j = 0; j < 8; ++j) out[(size_t)n * D_MODEL + j * 256 + t] = acc[j];
}

// ---------------------------------------------------------------------------
extern "C" void kernel_launch(void* const* d_in, const int* in_sizes, int n_in,
                              void* d_out, int out_size, void* d_ws, size_t ws_size,
                              hipStream_t stream) {
    const float* x     = (const float*)d_in[0];
    const float* b_pre = (const float*)d_in[1];
    const float* W_enc = (const float*)d_in[2];
    const float* b_enc = (const float*)d_in[3];
    const float* W_dec = (const float*)d_in[4];
    const float* b_dec = (const float*)d_in[5];

    float* out = (float*)d_out;                              // [N_TOK, D_MODEL]
    float* z   = out + (size_t)N_TOK * D_MODEL;              // [N_TOK, D_SAE] (z_sparse slot)

    // workspace layout (~440 MB): xc bf16 | W_enc bf16 | W_dec^T f32 | pairs
    char* w = (char*)d_ws;
    unsigned short* xc  = (unsigned short*)w; w += (size_t)N_TOK * D_MODEL * 2;
    unsigned short* Wb  = (unsigned short*)w; w += (size_t)D_SAE * D_MODEL * 2;
    float*          WdT = (float*)w;          w += (size_t)D_SAE * D_MODEL * 4;
    int*            pidx = (int*)w;           w += (size_t)N_TOK * TOPK * 4;
    float*          pval = (float*)w;

    const int topk_smem = (D_SAE + 256) * sizeof(unsigned);  // 132 KB
    hipFuncSetAttribute((const void*)sae_topk,
                        hipFuncAttributeMaxDynamicSharedMemorySize, topk_smem);

    sae_prep_x<<<2048, 256, 0, stream>>>(x, b_pre, xc);
    sae_prep_w<<<4096, 256, 0, stream>>>(W_enc, Wb, D_SAE * D_MODEL);
    sae_transpose_wdec<<<dim3(D_SAE / 32, D_MODEL / 32), 256, 0, stream>>>(W_dec, WdT);
    sae_encode<<<dim3(D_SAE / 128, N_TOK / 128), 256, 0, stream>>>(xc, Wb, b_enc, z);
    sae_topk<<<N_TOK, 256, topk_smem, stream>>>(z, pidx, pval);
    sae_decode<<<N_TOK, 256, 0, stream>>>(WdT, pidx, pval, b_dec, out);
}
